// WorldModel_85126251807477
// MI455X (gfx1250) — matrix operand order repaired and checked by reading, not compiled, not run on hardware
//
#include <hip/hip_runtime.h>

// ---------------------------------------------------------------------------
// RSSM rollout for MI455X (gfx1250): bf16 WMMA GEMMs (M=16 batch tile),
// weights pre-converted to bf16 + WMMA-tiled once per launch (fits in 192MB L2),
// one-hot straight-through sampling reduced to argmax + row gather.
// GEMM inner loop: 4x global_load_b128 + v_wmma, no LDS, no barriers.
// ---------------------------------------------------------------------------

typedef __attribute__((ext_vector_type(16))) __bf16 v16bf;
typedef __attribute__((ext_vector_type(8)))  __bf16 v8bf;
typedef __attribute__((ext_vector_type(8)))  float  v8f;

#define B_ 16
#define T_ 64
#define D_ 4096
#define H_ 1024
#define SC_ 1024
#define TOK_ 4096
#define FEAT_ 5120   // D + SC

// ---------------------------------------------------------------------------
// Weight conversion: fp32 row-major [K,N] -> bf16, tiled per (ntile, ktile)
// 32x16 B-fragment: within a tile, lane L owns column n=L&15, K-half L>>4,
// element e -> k = (L>>4)*8 + (e&7) + (e>>3)*16.  512 bf16 per tile, lane-major contiguous.
// Tile order: nt-major (so the GEMM k-loop walks contiguous tiles).
// ---------------------------------------------------------------------------
__global__ void conv_w_k(const float* __restrict__ src, __bf16* __restrict__ dst,
                         int K, int N) {
    size_t tid = (size_t)blockIdx.x * 256 + threadIdx.x;
    size_t total = (size_t)K * N;
    if (tid >= total) return;
    int nK = K >> 5;
    size_t tile = tid >> 9;
    int r = (int)(tid & 511);
    int lane = r >> 4, e = r & 15;
    int kt = (int)(tile % nK), nt = (int)(tile / nK);
    int n = (nt << 4) + (lane & 15);
    int k = (kt << 5) + ((lane >> 4) << 3) + (e & 7) + ((e >> 3) << 4);
    dst[tid] = (__bf16)src[(size_t)k * N + n];
}

// ---------------------------------------------------------------------------
// GEMM: out[16,N] = A[16,K](bf16,row-major,lda) @ W(tiled bf16) + bias
// blockDim = 256 (8 waves); each wave owns one 16-wide N tile.
// grid.x = (N/16)/8.  K multiple of 32, A 16B-aligned rows (lda multiple of 8).
//
// A fragment (16-bit 16x32 layout): lane (m = lane&15, half = lane>>4)
// holds K = half*8 + {0..7} and K = 16 + half*8 + {0..7}  -> two contiguous
// 16-byte global loads per k-tile; no LDS staging needed.
// ---------------------------------------------------------------------------
__global__ void gemm_wmma_k(const __bf16* __restrict__ A, int lda,
                            const __bf16* __restrict__ Wt,
                            const float* __restrict__ bias,
                            float* __restrict__ out, int ldout,
                            int K, int N) {
    const int lane = threadIdx.x & 31;
    const int wave = threadIdx.x >> 5;
    const int ntile = blockIdx.x * 8 + wave;
    const int nK = K >> 5;
    const int m = lane & 15, half = lane >> 4;

    const __bf16* arow = A + m * lda + (half << 3);
    const __bf16* bptr = Wt + (((size_t)ntile * nK) << 9) + (lane << 4);

    v8f acc = {};
#pragma unroll 2
    for (int kt = 0; kt < nK; ++kt) {
        v8bf alo = *(const v8bf*)(arow + (kt << 5));
        v8bf ahi = *(const v8bf*)(arow + (kt << 5) + 16);
        v16bf afrag = __builtin_shufflevector(alo, ahi,
            0, 1, 2, 3, 4, 5, 6, 7, 8, 9, 10, 11, 12, 13, 14, 15);
        v16bf bfrag = *(const v16bf*)(bptr + ((size_t)kt << 9));
        acc = __builtin_amdgcn_wmma_f32_16x16x32_bf16(
            false, afrag, false, bfrag, (short)0, acc, false, false);
    }
    // D layout: vgpr r, lane L -> row = r + (L>>4)*8, col = L&15
    const int col = (ntile << 4) + m;
    const float bcol = bias[col];
#pragma unroll
    for (int r = 0; r < 8; ++r) {
        int row = r + half * 8;
        out[row * ldout + col] = acc[r] + bcol;
    }
}

// ---------------------------------------------------------------------------
// LayerNorm + SiLU over H=1024 features, one block (256 thr) per batch row.
// Writes bf16 with arbitrary row stride (lets us write into concat buffers).
// ---------------------------------------------------------------------------
__global__ void ln_silu_k(const float* __restrict__ in, int ldin,
                          const float* __restrict__ g, const float* __restrict__ b,
                          __bf16* __restrict__ out, int ldout) {
    __shared__ float sd[256];
    int row = blockIdx.x, tid = threadIdx.x;
    float v[4]; float s = 0.f;
#pragma unroll
    for (int j = 0; j < 4; ++j) { v[j] = in[row * ldin + tid + (j << 8)]; s += v[j]; }
    sd[tid] = s; __syncthreads();
    for (int off = 128; off > 0; off >>= 1) {
        if (tid < off) sd[tid] += sd[tid + off];
        __syncthreads();
    }
    float mean = sd[0] * (1.f / 1024.f);
    __syncthreads();
    float vs = 0.f;
#pragma unroll
    for (int j = 0; j < 4; ++j) { float d = v[j] - mean; vs += d * d; }
    sd[tid] = vs; __syncthreads();
    for (int off = 128; off > 0; off >>= 1) {
        if (tid < off) sd[tid] += sd[tid + off];
        __syncthreads();
    }
    float rstd = rsqrtf(sd[0] * (1.f / 1024.f) + 1e-5f);
#pragma unroll
    for (int j = 0; j < 4; ++j) {
        int h = tid + (j << 8);
        float y = (v[j] - mean) * rstd * g[h] + b[h];
        float si = y / (1.f + expf(-y));
        out[row * ldout + h] = (__bf16)si;
    }
}

// ---------------------------------------------------------------------------
// unimix-categorical gumbel-max argmax over C=32 (fwd of straight-through)
// ---------------------------------------------------------------------------
__device__ __forceinline__ int unimix_argmax(const float* l, const float* g) {
    float mx = -3.4e38f;
#pragma unroll
    for (int c = 0; c < 32; ++c) mx = fmaxf(mx, l[c]);
    float sum = 0.f;
#pragma unroll
    for (int c = 0; c < 32; ++c) sum += expf(l[c] - mx);
    float inv = 0.99f / sum;
    float best = -3.4e38f; int bi = 0;
#pragma unroll
    for (int c = 0; c < 32; ++c) {
        float pr = expf(l[c] - mx) * inv + (0.01f / 32.0f);
        float sc = logf(pr) + g[c];
        if (sc > best) { best = sc; bi = c; }
    }
    return bi;
}

// posterior sample -> indices only (feeds the gW row-gather)
__global__ void post_sample_k(const float* __restrict__ logits,
                              const float* __restrict__ gum,
                              int* __restrict__ idx) {
    int p = blockIdx.x * 256 + threadIdx.x;   // 512 (b,s) pairs
    if (p >= 512) return;
    idx[p] = unimix_argmax(logits + p * 32, gum + p * 32);
}

// prior sample -> one-hot written straight into the output feature slab
__global__ void prior_sample_k(const float* __restrict__ logits,
                               const float* __restrict__ gum,
                               float* __restrict__ outp, int t) {
    int p = blockIdx.x * 256 + threadIdx.x;   // 512 (b,s) pairs
    if (p >= 512) return;
    int b = p >> 5, s = p & 31;
    int bi = unimix_argmax(logits + p * 32, gum + p * 32);
    float* o = outp + (((size_t)b * T_ + t) * FEAT_) + D_ + (s << 5);
#pragma unroll
    for (int c = 0; c < 32; ++c) o[c] = (c == bi) ? 1.0f : 0.0f;
}

// x_pre = onehot-gather of 32 gW rows + act @ gW[1024:1030] + gb   (fp32)
__global__ void gather_x_k(const int* __restrict__ idx,
                           const float* __restrict__ actions, int t,
                           const float* __restrict__ gW,
                           const float* __restrict__ gb,
                           float* __restrict__ x_pre) {
    int tid = blockIdx.x * 256 + threadIdx.x;   // 16 * 1024
    int b = tid >> 10, h = tid & 1023;
    float acc = gb[h];
    const int* ib = idx + b * 32;
#pragma unroll 4
    for (int s = 0; s < 32; ++s)
        acc += gW[(size_t)((s << 5) + ib[s]) * H_ + h];
    const float* ab = actions + ((size_t)b * T_ + t) * 6;
#pragma unroll
    for (int a = 0; a < 6; ++a)
        acc += ab[a] * gW[(size_t)(SC_ + a) * H_ + h];
    x_pre[tid] = acc;
}

// GRU gate update; also refreshes the bf16 deter copies inside the two
// pre-concatenated GEMM input buffers, and writes deter to the output slab.
__global__ void gru_update_k(const float* __restrict__ z_pre,
                             const float* __restrict__ c_pre,
                             float* __restrict__ deter,
                             __bf16* __restrict__ Apost,   // [16, 8192]
                             __bf16* __restrict__ hbuf,    // [16, 5120]
                             float* __restrict__ outp, int t) {
    int tid = blockIdx.x * 256 + threadIdx.x;   // 16 * 4096
    int b = tid >> 12, j = tid & 4095;
    float z = 1.f / (1.f + expf(-z_pre[tid]));
    float c = tanhf(c_pre[tid]);
    float d = (1.f - z) * deter[tid] + z * c;
    deter[tid] = d;
    __bf16 db = (__bf16)d;
    Apost[b * (TOK_ + D_) + TOK_ + j] = db;
    hbuf[b * FEAT_ + H_ + j] = db;
    outp[((size_t)b * T_ + t) * FEAT_ + j] = d;
}

// copy this step's token block (bf16) into the posterior GEMM A buffer
__global__ void tok_to_A_k(const float* __restrict__ tokens, int t,
                           __bf16* __restrict__ Apost) {
    int tid = blockIdx.x * 256 + threadIdx.x;   // 16 * 4096
    int b = tid >> 12, j = tid & 4095;
    Apost[b * (TOK_ + D_) + j] = (__bf16)tokens[((size_t)b * T_ + t) * TOK_ + j];
}

// deter0 = 0 (fp32 master + both bf16 copies)
__global__ void init_state_k(float* __restrict__ deter,
                             __bf16* __restrict__ Apost,
                             __bf16* __restrict__ hbuf) {
    int tid = blockIdx.x * 256 + threadIdx.x;   // 16 * 4096
    int b = tid >> 12, j = tid & 4095;
    deter[tid] = 0.f;
    Apost[b * (TOK_ + D_) + TOK_ + j] = (__bf16)0.f;
    hbuf[b * FEAT_ + H_ + j] = (__bf16)0.f;
}

// ---------------------------------------------------------------------------
extern "C" void kernel_launch(void* const* d_in, const int* in_sizes, int n_in,
                              void* d_out, int out_size, void* d_ws, size_t ws_size,
                              hipStream_t stream) {
    (void)in_sizes; (void)n_in; (void)out_size; (void)ws_size;
    const float* tokens  = (const float*)d_in[0];
    const float* actions = (const float*)d_in[1];
    const float* g_post  = (const float*)d_in[2];
    const float* g_prior = (const float*)d_in[3];
    const float* pW1=(const float*)d_in[4],  *pb1=(const float*)d_in[5],
               * pg1=(const float*)d_in[6],  *pB1=(const float*)d_in[7];
    const float* pW2=(const float*)d_in[8],  *pb2=(const float*)d_in[9],
               * pg2=(const float*)d_in[10], *pB2=(const float*)d_in[11];
    const float* pWo=(const float*)d_in[12], *pbo=(const float*)d_in[13];
    const float* qW1=(const float*)d_in[14], *qb1=(const float*)d_in[15],
               * qg1=(const float*)d_in[16], *qB1=(const float*)d_in[17];
    const float* qW2=(const float*)d_in[18], *qb2=(const float*)d_in[19],
               * qg2=(const float*)d_in[20], *qB2=(const float*)d_in[21];
    const float* qWo=(const float*)d_in[22], *qbo=(const float*)d_in[23];
    const float* gW =(const float*)d_in[24], *gb =(const float*)d_in[25],
               * gg =(const float*)d_in[26], *gB =(const float*)d_in[27];
    const float* gWz=(const float*)d_in[28], *gbz=(const float*)d_in[29];
    const float* gWc=(const float*)d_in[30], *gbc=(const float*)d_in[31];
    float* out = (float*)d_out;

    // workspace bump allocator (256B aligned)
    char* w = (char*)d_ws;
    auto alloc = [&](size_t bytes) -> char* {
        char* p = w; w += (bytes + 255) & ~(size_t)255; return p;
    };
    // bf16 WMMA-tiled weights (~117 MB total -> resident in 192 MB L2)
    __bf16* pW1t = (__bf16*)alloc((size_t)8192*1024*2);
    __bf16* pW2t = (__bf16*)alloc((size_t)1024*1024*2);
    __bf16* pWot = (__bf16*)alloc((size_t)1024*1024*2);
    __bf16* qW1t = (__bf16*)alloc((size_t)4096*1024*2);
    __bf16* qW2t = (__bf16*)alloc((size_t)1024*1024*2);
    __bf16* qWot = (__bf16*)alloc((size_t)1024*1024*2);
    __bf16* gWzt = (__bf16*)alloc((size_t)5120*4096*2);
    __bf16* gWct = (__bf16*)alloc((size_t)5120*4096*2);
    // activations
    __bf16* Apost = (__bf16*)alloc((size_t)16*8192*2);   // [tok | deter] bf16
    __bf16* hbuf  = (__bf16*)alloc((size_t)16*5120*2);   // [x   | deter] bf16
    __bf16* act1  = (__bf16*)alloc((size_t)16*1024*2);   // hidden bf16
    float*  deter = (float*) alloc((size_t)16*4096*4);
    float*  bufA  = (float*) alloc((size_t)16*1024*4);
    float*  bufB  = (float*) alloc((size_t)16*1024*4);
    float*  zpre  = (float*) alloc((size_t)16*4096*4);
    float*  cpre  = (float*) alloc((size_t)16*4096*4);
    int*    idxb  = (int*)   alloc((size_t)16*32*4);

    auto blocks = [](size_t n) { return (unsigned)((n + 255) / 256); };

    // one-time (per launch) fp32 -> bf16 tiled weight conversion
    conv_w_k<<<blocks((size_t)8192*1024), 256, 0, stream>>>(pW1, pW1t, 8192, 1024);
    conv_w_k<<<blocks((size_t)1024*1024), 256, 0, stream>>>(pW2, pW2t, 1024, 1024);
    conv_w_k<<<blocks((size_t)1024*1024), 256, 0, stream>>>(pWo, pWot, 1024, 1024);
    conv_w_k<<<blocks((size_t)4096*1024), 256, 0, stream>>>(qW1, qW1t, 4096, 1024);
    conv_w_k<<<blocks((size_t)1024*1024), 256, 0, stream>>>(qW2, qW2t, 1024, 1024);
    conv_w_k<<<blocks((size_t)1024*1024), 256, 0, stream>>>(qWo, qWot, 1024, 1024);
    conv_w_k<<<blocks((size_t)5120*4096), 256, 0, stream>>>(gWz, gWzt, 5120, 4096);
    conv_w_k<<<blocks((size_t)5120*4096), 256, 0, stream>>>(gWc, gWct, 5120, 4096);

    init_state_k<<<256, 256, 0, stream>>>(deter, Apost, hbuf);

    for (int t = 0; t < T_; ++t) {
        const float* gpt = g_post  + (size_t)t * 16 * 1024;
        const float* gqt = g_prior + (size_t)t * 16 * 1024;

        // posterior MLP on [tok | deter]
        tok_to_A_k<<<256, 256, 0, stream>>>(tokens, t, Apost);
        gemm_wmma_k<<<8, 256, 0, stream>>>(Apost, 8192, pW1t, pb1, bufA, 1024, 8192, 1024);
        ln_silu_k<<<16, 256, 0, stream>>>(bufA, 1024, pg1, pB1, act1, 1024);
        gemm_wmma_k<<<8, 256, 0, stream>>>(act1, 1024, pW2t, pb2, bufB, 1024, 1024, 1024);
        ln_silu_k<<<16, 256, 0, stream>>>(bufB, 1024, pg2, pB2, act1, 1024);
        gemm_wmma_k<<<8, 256, 0, stream>>>(act1, 1024, pWot, pbo, bufA, 1024, 1024, 1024);
        post_sample_k<<<2, 256, 0, stream>>>(bufA, gpt, idxb);

        // MiniGru: one-hot -> row gather, then two 5120x4096 gate GEMMs
        gather_x_k<<<64, 256, 0, stream>>>(idxb, actions, t, gW, gb, bufB);
        ln_silu_k<<<16, 256, 0, stream>>>(bufB, 1024, gg, gB, hbuf, 5120);
        gemm_wmma_k<<<32, 256, 0, stream>>>(hbuf, 5120, gWzt, gbz, zpre, 4096, 5120, 4096);
        gemm_wmma_k<<<32, 256, 0, stream>>>(hbuf, 5120, gWct, gbc, cpre, 4096, 5120, 4096);
        gru_update_k<<<256, 256, 0, stream>>>(zpre, cpre, deter, Apost, hbuf, out, t);

        // prior MLP on new deter (read via hbuf's deter slab, lda = 5120)
        gemm_wmma_k<<<8, 256, 0, stream>>>(hbuf + 1024, 5120, qW1t, qb1, bufA, 1024, 4096, 1024);
        ln_silu_k<<<16, 256, 0, stream>>>(bufA, 1024, qg1, qB1, act1, 1024);
        gemm_wmma_k<<<8, 256, 0, stream>>>(act1, 1024, qW2t, qb2, bufB, 1024, 1024, 1024);
        ln_silu_k<<<16, 256, 0, stream>>>(bufB, 1024, qg2, qB2, act1, 1024);
        gemm_wmma_k<<<8, 256, 0, stream>>>(act1, 1024, qWot, qbo, bufA, 1024, 1024, 1024);
        prior_sample_k<<<2, 256, 0, stream>>>(bufA, gqt, out, t);
    }
}
